// LlamaAttention_18279380812513
// MI455X (gfx1250) — compile-verified, matrix operand order checked by reading, not run on hardware
//
#include <hip/hip_runtime.h>
#include <hip/hip_bf16.h>
#include <math.h>

#define B_   2
#define L_   2048
#define HID_ 2048
#define NH_  32
#define NKV_ 8
#define HD_  64
#define MROWS_ (B_ * L_)   // 4096

typedef __bf16 v16bf __attribute__((ext_vector_type(16)));
typedef __bf16 v8bf  __attribute__((ext_vector_type(8)));
typedef float  v8f   __attribute__((ext_vector_type(8)));

union BF16x16 { v16bf v; v8bf h[2]; };

// Per-lane A/B operand fragment for v_wmma_*_16x16x32_bf16:
// elements p[0..7] and p[16..23] (two 16-byte loads), caller folds in
// row*ld + 8*(lane>>4).
__device__ __forceinline__ v16bf load_ab_tile(const __bf16* p) {
  BF16x16 u;
  u.h[0] = *reinterpret_cast<const v8bf*>(p);
  u.h[1] = *reinterpret_cast<const v8bf*>(p + 16);
  return u.v;
}

__device__ __forceinline__ v8f wmma_bf16(v16bf a, v16bf b, v8f c) {
  return __builtin_amdgcn_wmma_f32_16x16x32_bf16(false, a, false, b, (short)0, c,
                                                 false, false);
}

__device__ __forceinline__ float rowmax16(float x) {
  x = fmaxf(x, __shfl_xor(x, 1, 32));
  x = fmaxf(x, __shfl_xor(x, 2, 32));
  x = fmaxf(x, __shfl_xor(x, 4, 32));
  x = fmaxf(x, __shfl_xor(x, 8, 32));
  return x;
}
__device__ __forceinline__ float rowsum16(float x) {
  x += __shfl_xor(x, 1, 32);
  x += __shfl_xor(x, 2, 32);
  x += __shfl_xor(x, 4, 32);
  x += __shfl_xor(x, 8, 32);
  return x;
}

// ---------------- prep kernels ----------------

__global__ void cvt_bf16_kernel(const float* __restrict__ src,
                                __bf16* __restrict__ dst, int n) {
  int i = blockIdx.x * blockDim.x + threadIdx.x;
  if (i < n) dst[i] = (__bf16)src[i];
}

// W [K x N] (row-major) -> T [N x K] bf16 (row-major)
__global__ void transpose_bf16_kernel(const float* __restrict__ W,
                                      __bf16* __restrict__ T, int K, int N) {
  int i = blockIdx.x * blockDim.x + threadIdx.x;
  if (i < K * N) {
    int k = i / N;
    int n = i - k * N;
    T[(size_t)n * K + k] = (__bf16)W[i];
  }
}

// ---------------- QKV projection + RoPE ----------------
// grid: (MROWS/64, NH+2*NKV), block: 32. One wave computes a 64x64 tile:
// four 16-row A tiles share every B (weight) fragment.
// __launch_bounds__(32, 1): allow full VGPR budget, no spills.
__global__ void __launch_bounds__(32, 1)
qkv_rope_kernel(const __bf16* __restrict__ hb,
                const __bf16* __restrict__ WqT,
                const __bf16* __restrict__ WkT,
                const __bf16* __restrict__ WvT,
                __bf16* __restrict__ Qr,
                __bf16* __restrict__ Kr,
                __bf16* __restrict__ Vt) {
  const int lane = threadIdx.x;
  const int half = lane >> 4;
  const int lm   = lane & 15;
  const int r0   = blockIdx.x * 64;
  const int hu   = blockIdx.y;

  const __bf16* Bt;
  int head, mode;
  if (hu < NH_)              { Bt = WqT; head = hu;              mode = 0; }
  else if (hu < NH_ + NKV_)  { Bt = WkT; head = hu - NH_;        mode = 1; }
  else                       { Bt = WvT; head = hu - NH_ - NKV_; mode = 2; }
  const int n0 = head * HD_;

  v8f acc[4][4];
  #pragma unroll
  for (int t = 0; t < 4; ++t)
    #pragma unroll
    for (int cb = 0; cb < 4; ++cb) acc[t][cb] = {};

  const __bf16* arow = hb + (size_t)(r0 + lm) * HID_ + half * 8;
  const __bf16* brow = Bt + (size_t)(n0 + lm) * HID_ + half * 8;

  for (int k0 = 0; k0 < HID_; k0 += 32) {
    v16bf a[4];
    #pragma unroll
    for (int t = 0; t < 4; ++t)
      a[t] = load_ab_tile(arow + (size_t)(t * 16) * HID_ + k0);
    #pragma unroll
    for (int cb = 0; cb < 4; ++cb) {
      v16bf b = load_ab_tile(brow + (size_t)(cb * 16) * HID_ + k0);
      #pragma unroll
      for (int t = 0; t < 4; ++t) acc[t][cb] = wmma_bf16(a[t], b, acc[t][cb]);
    }
  }

  if (mode < 2) {
    // RoPE: feature d pairs with d+32 -> acc block cb pairs with cb+2,
    // same lane, same accumulator slot.
    #pragma unroll
    for (int t = 0; t < 4; ++t) {
      #pragma unroll
      for (int cb = 0; cb < 2; ++cb) {
        const int d = cb * 16 + lm;
        const float inv_freq = __expf(-0.28782313662425574f * (float)d); // theta^(-d/32)
        #pragma unroll
        for (int v = 0; v < 8; ++v) {
          const int row = r0 + t * 16 + v + 8 * half;
          const int p   = row & (L_ - 1);
          float s, c;
          __sincosf((float)p * inv_freq, &s, &c);
          const float x1 = acc[t][cb][v], x2 = acc[t][cb + 2][v];
          acc[t][cb][v]     = x1 * c - x2 * s;
          acc[t][cb + 2][v] = x2 * c + x1 * s;
        }
      }
    }
  }

  #pragma unroll
  for (int t = 0; t < 4; ++t) {
    #pragma unroll
    for (int cb = 0; cb < 4; ++cb) {
      const int d = cb * 16 + lm;
      #pragma unroll
      for (int v = 0; v < 8; ++v) {
        const int row = r0 + t * 16 + v + 8 * half;
        const int bb  = row >> 11;        // / L_
        const int p   = row & (L_ - 1);
        const __bf16 val = (__bf16)acc[t][cb][v];
        if (mode == 0)
          Qr[(((size_t)bb * NH_ + head) * L_ + p) * HD_ + d] = val;
        else if (mode == 1)
          Kr[(((size_t)bb * NKV_ + head) * L_ + p) * HD_ + d] = val;
        else
          Vt[(((size_t)bb * NKV_ + head) * HD_ + d) * L_ + p] = val;   // V^T
      }
    }
  }
}

// ---------------- causal flash attention (GQA) ----------------
// grid: (L/32, NH, B), block: 32. One wave owns 32 queries x full head dim:
// two Q tiles share every K and V fragment.
__global__ void __launch_bounds__(32, 1)
flash_attn_kernel(const __bf16* __restrict__ Qr,
                  const __bf16* __restrict__ Kr,
                  const __bf16* __restrict__ Vt,
                  __bf16* __restrict__ AO) {
  __shared__ alignas(16) __bf16 pt[2][16 * 32];  // P tiles: C-layout -> A-layout
  const int lane = threadIdx.x;
  const int half = lane >> 4;
  const int lm   = lane & 15;
  const int r0   = blockIdx.x * 32;
  const int h    = blockIdx.y;
  const int b    = blockIdx.z;
  const int kvh  = h >> 2;                       // group = NH/NKV = 4

  v16bf qa[2][2];
  #pragma unroll
  for (int t = 0; t < 2; ++t) {
    const __bf16* qp =
        Qr + (((size_t)b * NH_ + h) * L_ + r0 + t * 16 + lm) * HD_ + half * 8;
    #pragma unroll
    for (int c = 0; c < 2; ++c) qa[t][c] = load_ab_tile(qp + c * 32);
  }

  const __bf16* kbase = Kr + ((size_t)b * NKV_ + kvh) * L_ * HD_;
  const __bf16* vbase = Vt + ((size_t)b * NKV_ + kvh) * (size_t)HD_ * L_;

  float mrow[2][8], lrow[2][8];
  v8f o[2][4];
  #pragma unroll
  for (int t = 0; t < 2; ++t) {
    #pragma unroll
    for (int v = 0; v < 8; ++v) { mrow[t][v] = -INFINITY; lrow[t][v] = 0.0f; }
    #pragma unroll
    for (int cb = 0; cb < 4; ++cb) o[t][cb] = {};
  }

  for (int j0 = 0; j0 < r0 + 32; j0 += 32) {     // causal: keys [0, r0+32)
    v8f s[2][2];
    s[0][0] = {}; s[0][1] = {}; s[1][0] = {}; s[1][1] = {};
    #pragma unroll
    for (int jb = 0; jb < 2; ++jb) {
      const __bf16* kp = kbase + (size_t)(j0 + jb * 16 + lm) * HD_ + half * 8;
      #pragma unroll
      for (int c = 0; c < 2; ++c) {
        v16bf kt = load_ab_tile(kp + c * 32);
        s[0][jb] = wmma_bf16(qa[0][c], kt, s[0][jb]);
        s[1][jb] = wmma_bf16(qa[1][c], kt, s[1][jb]);
      }
    }

    #pragma unroll
    for (int t = 0; t < 2; ++t) {
      #pragma unroll
      for (int v = 0; v < 8; ++v) {
        const int q = r0 + t * 16 + v + 8 * half;
        float s0 = s[t][0][v] * 0.125f;          // 1/sqrt(64)
        float s1 = s[t][1][v] * 0.125f;
        if (j0 + lm      > q) s0 = -INFINITY;
        if (j0 + 16 + lm > q) s1 = -INFINITY;
        const float rv   = rowmax16(fmaxf(s0, s1));
        const float mnew = fmaxf(mrow[t][v], rv);
        const float corr = __expf(mrow[t][v] - mnew);
        mrow[t][v] = mnew;
        const float p0 = __expf(s0 - mnew);
        const float p1 = __expf(s1 - mnew);
        lrow[t][v] = lrow[t][v] * corr + rowsum16(p0 + p1);
        #pragma unroll
        for (int cb = 0; cb < 4; ++cb) o[t][cb][v] *= corr;
        pt[t][(v + 8 * half) * 32 + lm]      = (__bf16)p0;
        pt[t][(v + 8 * half) * 32 + 16 + lm] = (__bf16)p1;
      }
    }
    __syncthreads();
    v16bf pa[2];
    #pragma unroll
    for (int t = 0; t < 2; ++t)
      pa[t] = load_ab_tile(&pt[t][lm * 32 + half * 8]);   // A-layout readback
    #pragma unroll
    for (int cb = 0; cb < 4; ++cb) {
      const __bf16* vp = vbase + (size_t)(cb * 16 + lm) * L_ + j0 + half * 8;
      v16bf vv = load_ab_tile(vp);
      o[0][cb] = wmma_bf16(pa[0], vv, o[0][cb]);
      o[1][cb] = wmma_bf16(pa[1], vv, o[1][cb]);
    }
    __syncthreads();
  }

  #pragma unroll
  for (int t = 0; t < 2; ++t)
    #pragma unroll
    for (int cb = 0; cb < 4; ++cb)
      #pragma unroll
      for (int v = 0; v < 8; ++v) {
        const int q = r0 + t * 16 + v + 8 * half;
        AO[((size_t)(b * L_ + q)) * (NH_ * HD_) + h * HD_ + cb * 16 + lm] =
            (__bf16)(o[t][cb][v] / lrow[t][v]);
      }
}

// ---------------- output projection ----------------
// grid: (MROWS/64, HID/64), block: 32. 64x64 tile per wave.
__global__ void __launch_bounds__(32, 1)
oproj_kernel(const __bf16* __restrict__ AO,
             const __bf16* __restrict__ WoT,
             float* __restrict__ out) {
  const int lane = threadIdx.x;
  const int half = lane >> 4;
  const int lm   = lane & 15;
  const int r0   = blockIdx.x * 64;
  const int n0   = blockIdx.y * 64;

  v8f acc[4][4];
  #pragma unroll
  for (int t = 0; t < 4; ++t)
    #pragma unroll
    for (int cb = 0; cb < 4; ++cb) acc[t][cb] = {};

  const __bf16* arow = AO  + (size_t)(r0 + lm) * HID_ + half * 8;
  const __bf16* brow = WoT + (size_t)(n0 + lm) * HID_ + half * 8;

  for (int k0 = 0; k0 < HID_; k0 += 32) {
    v16bf a[4];
    #pragma unroll
    for (int t = 0; t < 4; ++t)
      a[t] = load_ab_tile(arow + (size_t)(t * 16) * HID_ + k0);
    #pragma unroll
    for (int cb = 0; cb < 4; ++cb) {
      v16bf b = load_ab_tile(brow + (size_t)(cb * 16) * HID_ + k0);
      #pragma unroll
      for (int t = 0; t < 4; ++t) acc[t][cb] = wmma_bf16(a[t], b, acc[t][cb]);
    }
  }

  #pragma unroll
  for (int t = 0; t < 4; ++t)
    #pragma unroll
    for (int cb = 0; cb < 4; ++cb)
      #pragma unroll
      for (int v = 0; v < 8; ++v)
        out[(size_t)(r0 + t * 16 + v + 8 * half) * HID_ + n0 + cb * 16 + lm] =
            acc[t][cb][v];
}

// ---------------- host launch ----------------

extern "C" void kernel_launch(void* const* d_in, const int* in_sizes, int n_in,
                              void* d_out, int out_size, void* d_ws, size_t ws_size,
                              hipStream_t stream) {
  (void)in_sizes; (void)n_in; (void)out_size; (void)ws_size;
  const float* hidden = (const float*)d_in[0];
  const float* Wq = (const float*)d_in[1];
  const float* Wk = (const float*)d_in[2];
  const float* Wv = (const float*)d_in[3];
  const float* Wo = (const float*)d_in[4];
  float* out = (float*)d_out;

  __bf16* p = (__bf16*)d_ws;
  __bf16* hb  = p; p += (size_t)MROWS_ * HID_;          // hidden bf16 [4096,2048]
  __bf16* WqT = p; p += (size_t)HID_ * HID_;            // Wq^T  [2048,2048]
  __bf16* WkT = p; p += (size_t)(NKV_ * HD_) * HID_;    // Wk^T  [512,2048]
  __bf16* WvT = p; p += (size_t)(NKV_ * HD_) * HID_;    // Wv^T  [512,2048]
  __bf16* WoT = p; p += (size_t)HID_ * HID_;            // Wo^T  [2048,2048]
  __bf16* Qr  = p; p += (size_t)B_ * NH_ * L_ * HD_;    // Q rope'd [B,NH,L,64]
  __bf16* Kr  = p; p += (size_t)B_ * NKV_ * L_ * HD_;   // K rope'd [B,NKV,L,64]
  __bf16* Vt  = p; p += (size_t)B_ * NKV_ * HD_ * L_;   // V^T      [B,NKV,64,L]
  __bf16* AO  = p; p += (size_t)MROWS_ * HID_;          // attn out [4096,2048]

  int n = MROWS_ * HID_;
  cvt_bf16_kernel<<<(n + 255) / 256, 256, 0, stream>>>(hidden, hb, n);
  n = HID_ * HID_;
  transpose_bf16_kernel<<<(n + 255) / 256, 256, 0, stream>>>(Wq, WqT, HID_, HID_);
  n = HID_ * (NKV_ * HD_);
  transpose_bf16_kernel<<<(n + 255) / 256, 256, 0, stream>>>(Wk, WkT, HID_, NKV_ * HD_);
  transpose_bf16_kernel<<<(n + 255) / 256, 256, 0, stream>>>(Wv, WvT, HID_, NKV_ * HD_);
  n = HID_ * HID_;
  transpose_bf16_kernel<<<(n + 255) / 256, 256, 0, stream>>>(Wo, WoT, HID_, HID_);

  qkv_rope_kernel<<<dim3(MROWS_ / 64, NH_ + 2 * NKV_), 32, 0, stream>>>(
      hb, WqT, WkT, WvT, Qr, Kr, Vt);
  flash_attn_kernel<<<dim3(L_ / 32, NH_, B_), 32, 0, stream>>>(Qr, Kr, Vt, AO);
  oproj_kernel<<<dim3(MROWS_ / 64, HID_ / 64), 32, 0, stream>>>(AO, WoT, out);
}
